// PRCLayer_82729660056158
// MI455X (gfx1250) — matrix operand order repaired and checked by reading, not compiled, not run on hardware
//
#include <hip/hip_runtime.h>
#include <hip/hip_bf16.h>
#include <math.h>

// ---------------------------------------------------------------------------
// CDNA5 (gfx1250) implementation of the PRL-transformer block.
// wave32 everywhere; matrix math via v_wmma_f32_16x16x32_bf16.
// ---------------------------------------------------------------------------

typedef __bf16 bf16_t;
typedef __attribute__((ext_vector_type(16))) __bf16 v16bf;
typedef __attribute__((ext_vector_type(8)))  __bf16 v8bf;
typedef __attribute__((ext_vector_type(8)))  float  v8f;

static constexpr int kB   = 2;
static constexpr int kT   = 1024;
static constexpr int kD   = 1024;
static constexpr int kFF  = 4096;
static constexpr int kNH  = 16;
static constexpr int kHD  = 64;
static constexpr int kNP  = 32;    // experts / prototypes
static constexpr int kR   = 16;    // low rank == one WMMA tile dimension
static constexpr int kN   = kB * kT;   // 2048 tokens
static constexpr int kCAP = kN;        // max tokens per expert (top-2, distinct)

#define DEVI static __device__ __forceinline__

// ---- cross-lane helpers (wave32) ------------------------------------------
DEVI float red_max16(float v) {            // reduce over 16-lane half
#pragma unroll
  for (int m = 1; m < 16; m <<= 1) v = fmaxf(v, __shfl_xor(v, m, 32));
  return v;
}
DEVI float red_sum16(float v) {
#pragma unroll
  for (int m = 1; m < 16; m <<= 1) v += __shfl_xor(v, m, 32);
  return v;
}
DEVI float red_sum32(float v) {
#pragma unroll
  for (int m = 1; m < 32; m <<= 1) v += __shfl_xor(v, m, 32);
  return v;
}

// ---- WMMA wrapper ----------------------------------------------------------
DEVI v8f wmma_bf16(v16bf a, v16bf b, v8f c) {
  return __builtin_amdgcn_wmma_f32_16x16x32_bf16(
      /*neg_a=*/false, a, /*neg_b=*/false, b,
      /*c_mod=*/(short)0, c, /*reuse_a=*/false, /*reuse_b=*/false);
}

// A-fragment (16x32 bf16). Lane (half = lane>>4) holds row m = lane&15,
// elements 0..7  -> K = koff + 16*half + j
// elements 8..15 -> K = koff + 16*half + 8 + j
// rowPtr must already include the koff offset; 16B alignment guaranteed by callers.
DEVI v16bf load_a_frag(const bf16_t* rowPtr, int half) {
  const v8bf lo = *(const v8bf*)(rowPtr + 16 * half);
  const v8bf hi = *(const v8bf*)(rowPtr + 16 * half + 8);
  v16bf a;
#pragma unroll
  for (int e = 0; e < 8; ++e) { a[e] = lo[e]; a[e + 8] = hi[e]; }
  return a;
}

// ---------------------------------------------------------------------------
// small utility kernels
// ---------------------------------------------------------------------------
__global__ void k_f32_to_bf16(const float* __restrict__ in,
                              bf16_t* __restrict__ out, int n) {
  int i = blockIdx.x * blockDim.x + threadIdx.x;
  if (i < n) out[i] = (bf16_t)in[i];
}

__global__ void k_zero_i32(int* __restrict__ p, int n) {
  int i = blockIdx.x * blockDim.x + threadIdx.x;
  if (i < n) p[i] = 0;
}

// rmsnorm over last dim; writes f32 (for routing) + bf16 (for WMMA A-matrix)
__global__ void k_rmsnorm(const float* __restrict__ x, const float* __restrict__ w,
                          float* __restrict__ of, bf16_t* __restrict__ ob, int dim) {
  const int tok = blockIdx.x;
  const float* xr = x + (size_t)tok * dim;
  float s = 0.f;
  for (int i = threadIdx.x; i < dim; i += blockDim.x) { float v = xr[i]; s += v * v; }
  s = red_sum32(s);
  __shared__ float red[8];
  const int wid = threadIdx.x >> 5;
  if ((threadIdx.x & 31) == 0) red[wid] = s;
  __syncthreads();
  if (threadIdx.x == 0) {
    float t = 0.f;
    for (int i = 0; i < (int)(blockDim.x >> 5); ++i) t += red[i];
    red[0] = t;
  }
  __syncthreads();
  const float rs = rsqrtf(red[0] / (float)dim + 1e-6f);
  for (int i = threadIdx.x; i < dim; i += blockDim.x) {
    const float h = xr[i] * rs * w[i];
    of[(size_t)tok * dim + i] = h;
    ob[(size_t)tok * dim + i] = (bf16_t)h;
  }
}

// ---------------------------------------------------------------------------
// PRL routing: one wave per token, lane == prototype (NP == 32 == wave32).
// Computes top-2 experts + renormalized weights, scatters assignments.
// ---------------------------------------------------------------------------
__global__ void __launch_bounds__(32) k_prl_route(
    const float* __restrict__ x, const float* __restrict__ protos,
    const float* __restrict__ temp_ptr, int in_dim,
    int* __restrict__ counts, int* __restrict__ asgn_idx,
    float* __restrict__ asgn_w) {
  __shared__ float sx[kFF];   // max in_dim = 4096 -> 16 KB LDS
  const int tok = blockIdx.x;
  const int lane = threadIdx.x;
  const float* xr = x + (size_t)tok * in_dim;
  float x2 = 0.f;
  for (int i = lane; i < in_dim; i += 32) { float v = xr[i]; sx[i] = v; x2 += v * v; }
  x2 = red_sum32(x2);
  __syncthreads();

  const float* pr = protos + (size_t)lane * in_dim;
  float dot = 0.f, p2 = 0.f;
  for (int i = 0; i < in_dim; ++i) { const float p = pr[i]; dot += p * sx[i]; p2 += p * p; }
  const float d2   = x2 + p2 - 2.f * dot;
  const float dist = sqrtf(fmaxf(d2, 0.f));
  const float temp = fmaxf(fabsf(temp_ptr[0]), 0.1f);
  const float s    = -dist / temp;

  // top-1
  float b0 = s; int i0 = lane;
#pragma unroll
  for (int m = 16; m >= 1; m >>= 1) {
    float o = __shfl_xor(b0, m, 32); int oi = __shfl_xor(i0, m, 32);
    if (o > b0 || (o == b0 && oi < i0)) { b0 = o; i0 = oi; }
  }
  // top-2 (exclude i0)
  float b1 = (lane == i0) ? -3.4e38f : s; int i1 = lane;
#pragma unroll
  for (int m = 16; m >= 1; m >>= 1) {
    float o = __shfl_xor(b1, m, 32); int oi = __shfl_xor(i1, m, 32);
    if (o > b1 || (o == b1 && oi < i1)) { b1 = o; i1 = oi; }
  }
  // softmax denominator cancels under top-2 renormalization:
  const float w0 = 1.f / (1.f + __expf(b1 - b0));
  const float w1 = 1.f - w0;

  if (lane == 0) {
    int p0 = atomicAdd(&counts[i0], 1);
    asgn_idx[i0 * kCAP + p0] = (tok << 1);
    asgn_w  [i0 * kCAP + p0] = w0;
    int p1 = atomicAdd(&counts[i1], 1);
    asgn_idx[i1 * kCAP + p1] = (tok << 1) | 1;
    asgn_w  [i1 * kCAP + p1] = w1;
  }
}

// ---------------------------------------------------------------------------
// PRL grouped low-rank GEMM. One wave per (expert, 16-token tile).
// Stage 1: H(16 tok x 16 rank) = X_gathered(16 x in) * B_e^T  -> WMMA K-loop
// Stage 2: Y(16 tok x out)     = H * A_e^T (K=16, zero-padded to 32)
// Output written weight-scaled into per-slot buffers (no atomics needed).
// ---------------------------------------------------------------------------
__global__ void __launch_bounds__(32) k_prl_gemm(
    const bf16_t* __restrict__ Xbf,       // (N, in_dim)
    const bf16_t* __restrict__ Bw,        // (NP, R, in_dim)
    const bf16_t* __restrict__ Aw,        // (NP, out_dim, R)
    const float*  __restrict__ bias,      // (NP, out_dim)
    const int* __restrict__ counts,
    const int* __restrict__ asgn_idx, const float* __restrict__ asgn_w,
    float* __restrict__ y0, float* __restrict__ y1,
    int in_dim, int out_dim) {
  const int e    = blockIdx.y;
  const int tile = blockIdx.x;
  const int cnt  = counts[e];
  if (tile * 16 >= cnt) return;

  const int lane = threadIdx.x;
  const int half = lane >> 4;
  const int n    = lane & 15;

  __shared__ int   sTok[16];
  __shared__ int   sSlot[16];
  __shared__ float sW[16];
  __shared__ float sH[16 * 16];

  if (lane < 16) {
    const int pos = tile * 16 + lane;
    const int entry = (pos < cnt) ? asgn_idx[e * kCAP + pos] : asgn_idx[e * kCAP];
    sTok[lane]  = entry >> 1;
    sSlot[lane] = entry & 1;
    sW[lane]    = (pos < cnt) ? asgn_w[e * kCAP + pos] : 0.f;
  }
  __syncthreads();

  // ---- stage 1: H = X * B_e^T --------------------------------------------
  const bf16_t* xrow = Xbf + (size_t)sTok[n] * in_dim;                 // A row m = lane&15
  const bf16_t* brow = Bw + ((size_t)e * kR + n) * in_dim + 16 * half; // B col n, K offset
  v8f acc = {};
  for (int kk = 0; kk < in_dim; kk += 32) {
    v16bf a = load_a_frag(xrow + kk, half);
    v16bf b = *(const v16bf*)(brow + kk);    // 32B aligned contiguous K
    acc = wmma_bf16(a, b, acc);
  }
  // C layout -> LDS (rows m = v + 8*half, col n)
#pragma unroll
  for (int v = 0; v < 8; ++v) sH[(v + 8 * half) * 16 + n] = acc[v];
  __syncthreads();

  // H as A-fragment: row m = lane&15, K = rank (0..15), zero-pad K=16..31
  v16bf ha;
#pragma unroll
  for (int j = 0; j < 8; ++j) {
    ha[j]     = half ? (bf16_t)0.0f : (bf16_t)sH[n * 16 + j];
    ha[j + 8] = half ? (bf16_t)0.0f : (bf16_t)sH[n * 16 + 8 + j];
  }

  // ---- stage 2: Y = H * A_e^T, scale by routing weight, add bias ----------
  for (int n0 = 0; n0 < out_dim; n0 += 16) {
    const int col = n0 + n;
    v16bf bb;
    if (half == 0) {
      bb = *(const v16bf*)(Aw + ((size_t)e * out_dim + col) * kR);  // K = 0..15
    } else {
#pragma unroll
      for (int j = 0; j < 16; ++j) bb[j] = (bf16_t)0.0f;            // K = 16..31 padding
    }
    v8f y = {};
    y = wmma_bf16(ha, bb, y);
    const float bv = bias[(size_t)e * out_dim + col];
#pragma unroll
    for (int v = 0; v < 8; ++v) {
      const int m   = v + 8 * half;
      const int pos = tile * 16 + m;
      if (pos < cnt) {
        const float val = sW[m] * (y[v] + bv);
        float* dst = sSlot[m] ? y1 : y0;
        dst[(size_t)sTok[m] * out_dim + col] = val;
      }
    }
  }
}

// ---------------------------------------------------------------------------
// combines
// ---------------------------------------------------------------------------
// q/k/v: combine slots and pack into bf16 [b][h][t][hd] for the attention GEMMs
__global__ void k_combine_pack_qkv(const float* __restrict__ y0,
                                   const float* __restrict__ y1,
                                   bf16_t* __restrict__ pack) {
  const size_t i = (size_t)blockIdx.x * blockDim.x + threadIdx.x;
  if (i >= (size_t)kN * kD) return;
  const int tok = (int)(i / kD), c = (int)(i % kD);
  const int b = tok / kT, t = tok % kT, h = c / kHD, hd = c % kHD;
  const float val = y0[i] + y1[i];
  pack[(((size_t)b * kNH + h) * kT + t) * kHD + hd] = (bf16_t)val;
}

__global__ void k_combine_residual(const float* __restrict__ x,
                                   const float* __restrict__ y0,
                                   const float* __restrict__ y1,
                                   float* __restrict__ out, int n) {
  const int i = blockIdx.x * blockDim.x + threadIdx.x;
  if (i < n) out[i] = x[i] + y0[i] + y1[i];
}

__global__ void k_combine_sum(const float* __restrict__ y0,
                              const float* __restrict__ y1,
                              float* __restrict__ out, int n) {
  const int i = blockIdx.x * blockDim.x + threadIdx.x;
  if (i < n) out[i] = y0[i] + y1[i];
}

// ff = silu(gate) * (up0+up1); f32 for routing + bf16 for down-GEMM A-matrix
__global__ void k_combine_silu_mul(const float* __restrict__ gate,
                                   const float* __restrict__ y0,
                                   const float* __restrict__ y1,
                                   float* __restrict__ of,
                                   bf16_t* __restrict__ ob, int n) {
  const int i = blockIdx.x * blockDim.x + threadIdx.x;
  if (i < n) {
    const float g = gate[i];
    const float r = (g / (1.f + __expf(-g))) * (y0[i] + y1[i]);
    of[i] = r;
    ob[i] = (bf16_t)r;
  }
}

// ---------------------------------------------------------------------------
// Flash attention (causal). One wave per (b*NH, 16-row q tile).
// K-blocks of 32 keys; S via 2x2 WMMAs, online softmax, P*V via 4 WMMAs.
// q/k/v in bf16 [bh][T][HD]; output f32 (B,T,C) + bf16 copy for the o-PRL.
// ---------------------------------------------------------------------------
__global__ void __launch_bounds__(32) k_attn(
    const bf16_t* __restrict__ q, const bf16_t* __restrict__ k,
    const bf16_t* __restrict__ v,
    float* __restrict__ of, bf16_t* __restrict__ ob) {
  const int qt   = blockIdx.x;
  const int bh   = blockIdx.y;
  const int lane = threadIdx.x;
  const int half = lane >> 4;
  const int n    = lane & 15;
  const int q0   = qt * 16;
  const size_t base = (size_t)bh * kT * kHD;
  const float scale = 0.125f;   // 1/sqrt(64)

  // Q A-fragments (K-dim = HD = 64 -> two fragments)
  const bf16_t* qrow = q + base + (size_t)(q0 + n) * kHD;
  const v16bf qa0 = load_a_frag(qrow, half);
  const v16bf qa1 = load_a_frag(qrow + 32, half);

  float M[8], L[8];
  v8f o0 = {}, o1 = {}, o2 = {}, o3 = {};
#pragma unroll
  for (int vv = 0; vv < 8; ++vv) { M[vv] = -3.0e38f; L[vv] = 0.f; }

  __shared__ __align__(16) bf16_t sP[16 * 32];
  __shared__ __align__(16) bf16_t sV[32 * kHD];

  for (int j = 0; j <= q0 + 15; j += 32) {
    // prefetch next K block (speculative; global_prefetch_b8)
    if (j + 32 < kT) __builtin_prefetch(k + base + (size_t)(j + 32 + n) * kHD, 0, 0);

    // ---- S = Q K^T for two 16-key tiles ----
    const int kr0 = (j + n      < kT) ? (j + n)      : (kT - 1);
    const int kr1 = (j + 16 + n < kT) ? (j + 16 + n) : (kT - 1);
    const bf16_t* kp0 = k + base + (size_t)kr0 * kHD + 16 * half;
    const bf16_t* kp1 = k + base + (size_t)kr1 * kHD + 16 * half;
    v8f s0 = {}, s1 = {};
    s0 = wmma_bf16(qa0, *(const v16bf*)(kp0),      s0);
    s0 = wmma_bf16(qa1, *(const v16bf*)(kp0 + 32), s0);
    s1 = wmma_bf16(qa0, *(const v16bf*)(kp1),      s1);
    s1 = wmma_bf16(qa1, *(const v16bf*)(kp1 + 32), s1);

    // ---- online softmax over the 32-key block ----
#pragma unroll
    for (int vv = 0; vv < 8; ++vv) {
      const int row = q0 + vv + 8 * half;
      float x0 = s0[vv] * scale; if (j + n      > row) x0 = -1.0e30f;
      float x1 = s1[vv] * scale; if (j + 16 + n > row) x1 = -1.0e30f;
      float mm = red_max16(fmaxf(x0, x1));      // row max (stays within half)
      const float Mn = fmaxf(M[vv], mm);
      const float al = __expf(M[vv] - Mn);
      const float p0 = __expf(x0 - Mn);
      const float p1 = __expf(x1 - Mn);
      const float rs = red_sum16(p0 + p1);
      L[vv] = L[vv] * al + rs;
      M[vv] = Mn;
      o0[vv] *= al; o1[vv] *= al; o2[vv] *= al; o3[vv] *= al;
      sP[(vv + 8 * half) * 32 + n]      = (bf16_t)p0;
      sP[(vv + 8 * half) * 32 + 16 + n] = (bf16_t)p1;
    }
    __syncthreads();

    // ---- stage V block (32 x 64) into LDS, coalesced 128B per lane ----
    {
      const int vr = (j + lane < kT) ? (j + lane) : (kT - 1);
      const uint4* src = (const uint4*)(v + base + (size_t)vr * kHD);
      uint4* dst = (uint4*)(sV + lane * kHD);
#pragma unroll
      for (int i = 0; i < 8; ++i) dst[i] = src[i];
    }
    __syncthreads();

    // ---- O += P V ----
    v16bf pa;
#pragma unroll
    for (int jj = 0; jj < 8; ++jj) {
      pa[jj]     = sP[n * 32 + 16 * half + jj];
      pa[jj + 8] = sP[n * 32 + 16 * half + 8 + jj];
    }
#pragma unroll
    for (int tset = 0; tset < 4; ++tset) {
      const int n0 = tset * 16;
      v16bf vb;
#pragma unroll
      for (int e = 0; e < 16; ++e) vb[e] = sV[(16 * half + e) * kHD + n0 + n];
      if      (tset == 0) o0 = wmma_bf16(pa, vb, o0);
      else if (tset == 1) o1 = wmma_bf16(pa, vb, o1);
      else if (tset == 2) o2 = wmma_bf16(pa, vb, o2);
      else                o3 = wmma_bf16(pa, vb, o3);
    }
    __syncthreads();
  }

  // ---- finalize: /L, write (B,T,C) f32 + bf16 ----
  const int b = bh / kNH, h = bh % kNH;
#pragma unroll
  for (int vv = 0; vv < 8; ++vv) {
    const float inv = 1.f / L[vv];
    const int t = q0 + vv + 8 * half;
    const size_t row = ((size_t)b * kT + t) * kD + (size_t)h * kHD;
    const float r0 = o0[vv] * inv, r1 = o1[vv] * inv, r2 = o2[vv] * inv, r3 = o3[vv] * inv;
    of[row + 0  + n] = r0;  ob[row + 0  + n] = (bf16_t)r0;
    of[row + 16 + n] = r1;  ob[row + 16 + n] = (bf16_t)r1;
    of[row + 32 + n] = r2;  ob[row + 32 + n] = (bf16_t)r2;
    of[row + 48 + n] = r3;  ob[row + 48 + n] = (bf16_t)r3;
  }
}

// ---------------------------------------------------------------------------
// host orchestration
// ---------------------------------------------------------------------------
struct PrlParams { const float *proto, *Bw, *Aw, *bias, *temp; };

// Input dict flattening order detection: insertion {proto,B,A,bias,temp} vs
// pytree-alphabetical {A,B,bias,proto,temp}. First element size disambiguates
// (np*in != np*out*r for every layer here).
static PrlParams map_prl(void* const* d_in, const int* sz, int base,
                         long np, long in, long /*out*/, long /*r*/) {
  PrlParams P;
  if ((long)sz[base] == np * in) {
    P.proto = (const float*)d_in[base + 0];
    P.Bw    = (const float*)d_in[base + 1];
    P.Aw    = (const float*)d_in[base + 2];
    P.bias  = (const float*)d_in[base + 3];
    P.temp  = (const float*)d_in[base + 4];
  } else {
    P.Aw    = (const float*)d_in[base + 0];
    P.Bw    = (const float*)d_in[base + 1];
    P.bias  = (const float*)d_in[base + 2];
    P.proto = (const float*)d_in[base + 3];
    P.temp  = (const float*)d_in[base + 4];
  }
  return P;
}

extern "C" void kernel_launch(void* const* d_in, const int* in_sizes, int n_in,
                              void* d_out, int out_size, void* d_ws, size_t ws_size,
                              hipStream_t stream) {
  (void)n_in; (void)out_size;
  const float* x  = (const float*)d_in[0];
  PrlParams q_p    = map_prl(d_in, in_sizes, 1,  kNP, kD,  kD,  kR);
  PrlParams k_p    = map_prl(d_in, in_sizes, 6,  kNP, kD,  kD,  kR);
  PrlParams v_p    = map_prl(d_in, in_sizes, 11, kNP, kD,  kD,  kR);
  PrlParams o_p    = map_prl(d_in, in_sizes, 16, kNP, kD,  kD,  kR);
  PrlParams gate_p = map_prl(d_in, in_sizes, 21, kNP, kD,  kFF, kR);
  PrlParams up_p   = map_prl(d_in, in_sizes, 26, kNP, kD,  kFF, kR);
  PrlParams down_p = map_prl(d_in, in_sizes, 31, kNP, kFF, kD,  kR);
  const float* n1w = (const float*)d_in[36];
  const float* n2w = (const float*)d_in[37];

  // bump allocator over d_ws (256B aligned)
  char* wsb = (char*)d_ws; size_t off = 0; (void)ws_size;
  auto alloc = [&](size_t bytes) -> void* {
    size_t o = (off + 255) & ~(size_t)255; off = o + bytes; return wsb + o;
  };
  auto falloc = [&](size_t n) -> float*  { return (float*) alloc(n * sizeof(float)); };
  auto balloc = [&](size_t n) -> bf16_t* { return (bf16_t*)alloc(n * sizeof(bf16_t)); };

  const int TPB = 256;
  auto cvt = [&](const float* src, size_t n) -> bf16_t* {
    bf16_t* dst = balloc(n);
    k_f32_to_bf16<<<dim3((unsigned)((n + TPB - 1) / TPB)), TPB, 0, stream>>>(src, dst, (int)n);
    return dst;
  };

  // bf16 weight copies (L2-resident; converted every call -> deterministic)
  // NOTE: names must not shadow the k* shape constants (kB, kT, ...)
  bf16_t* qBw = cvt(q_p.Bw,    (size_t)kNP * kR * kD);
  bf16_t* qAw = cvt(q_p.Aw,    (size_t)kNP * kD * kR);
  bf16_t* kBw = cvt(k_p.Bw,    (size_t)kNP * kR * kD);
  bf16_t* kAw = cvt(k_p.Aw,    (size_t)kNP * kD * kR);
  bf16_t* vBw = cvt(v_p.Bw,    (size_t)kNP * kR * kD);
  bf16_t* vAw = cvt(v_p.Aw,    (size_t)kNP * kD * kR);
  bf16_t* oBw = cvt(o_p.Bw,    (size_t)kNP * kR * kD);
  bf16_t* oAw = cvt(o_p.Aw,    (size_t)kNP * kD * kR);
  bf16_t* gBw = cvt(gate_p.Bw, (size_t)kNP * kR * kD);
  bf16_t* gAw = cvt(gate_p.Aw, (size_t)kNP * kFF * kR);
  bf16_t* uBw = cvt(up_p.Bw,   (size_t)kNP * kR * kD);
  bf16_t* uAw = cvt(up_p.Aw,   (size_t)kNP * kFF * kR);
  bf16_t* dBw = cvt(down_p.Bw, (size_t)kNP * kR * kFF);
  bf16_t* dAw = cvt(down_p.Aw, (size_t)kNP * kD * kR);

  // activations / scratch
  float*  h1f   = falloc((size_t)kN * kD);
  bf16_t* h1b   = balloc((size_t)kN * kD);
  int*    counts = (int*)alloc(kNP * sizeof(int));
  int*    aidx   = (int*)alloc((size_t)kNP * kCAP * sizeof(int));
  float*  aw     = falloc((size_t)kNP * kCAP);
  float*  y0     = falloc((size_t)kN * kFF);
  float*  y1     = falloc((size_t)kN * kFF);
  bf16_t* qpk    = balloc((size_t)kN * kD);
  bf16_t* kpk    = balloc((size_t)kN * kD);
  bf16_t* vpk    = balloc((size_t)kN * kD);
  float*  attnf  = falloc((size_t)kN * kD);
  bf16_t* attnb  = balloc((size_t)kN * kD);
  float*  xmid   = falloc((size_t)kN * kD);
  float*  h2f    = falloc((size_t)kN * kD);
  bf16_t* h2b    = balloc((size_t)kN * kD);
  float*  gact   = falloc((size_t)kN * kFF);
  float*  fff    = falloc((size_t)kN * kFF);
  bf16_t* ffb    = balloc((size_t)kN * kFF);

  auto run_prl = [&](const PrlParams& P, const float* Xf, const bf16_t* Xbf,
                     const bf16_t* Bb, const bf16_t* Ab, int in_dim, int out_dim) {
    k_zero_i32<<<1, 32, 0, stream>>>(counts, kNP);
    k_prl_route<<<kN, 32, 0, stream>>>(Xf, P.proto, P.temp, in_dim, counts, aidx, aw);
    k_prl_gemm<<<dim3(kCAP / 16, kNP), 32, 0, stream>>>(
        Xbf, Bb, Ab, P.bias, counts, aidx, aw, y0, y1, in_dim, out_dim);
  };

  const int nD  = kN * kD;
  const int nFF = kN * kFF;
  const int gD  = (nD  + TPB - 1) / TPB;
  const int gFF = (nFF + TPB - 1) / TPB;

  // ---- block forward ------------------------------------------------------
  k_rmsnorm<<<kN, TPB, 0, stream>>>(x, n1w, h1f, h1b, kD);

  run_prl(q_p, h1f, h1b, qBw, qAw, kD, kD);
  k_combine_pack_qkv<<<gD, TPB, 0, stream>>>(y0, y1, qpk);
  run_prl(k_p, h1f, h1b, kBw, kAw, kD, kD);
  k_combine_pack_qkv<<<gD, TPB, 0, stream>>>(y0, y1, kpk);
  run_prl(v_p, h1f, h1b, vBw, vAw, kD, kD);
  k_combine_pack_qkv<<<gD, TPB, 0, stream>>>(y0, y1, vpk);

  k_attn<<<dim3(kT / 16, kB * kNH), 32, 0, stream>>>(qpk, kpk, vpk, attnf, attnb);

  run_prl(o_p, attnf, attnb, oBw, oAw, kD, kD);
  k_combine_residual<<<gD, TPB, 0, stream>>>(x, y0, y1, xmid, nD);

  k_rmsnorm<<<kN, TPB, 0, stream>>>(xmid, n2w, h2f, h2b, kD);

  run_prl(gate_p, h2f, h2b, gBw, gAw, kD, kFF);
  k_combine_sum<<<gFF, TPB, 0, stream>>>(y0, y1, gact, nFF);
  run_prl(up_p, h2f, h2b, uBw, uAw, kD, kFF);
  k_combine_silu_mul<<<gFF, TPB, 0, stream>>>(gact, y0, y1, fff, ffb, nFF);

  run_prl(down_p, fff, ffb, dBw, dAw, kFF, kD);
  k_combine_residual<<<gD, TPB, 0, stream>>>(xmid, y0, y1, (float*)d_out, nD);
}
